// IcoAttention_65678639891282
// MI455X (gfx1250) — matrix operand order repaired and checked by reading, not compiled, not run on hardware
//
#include <hip/hip_runtime.h>
#include <hip/hip_bf16.h>
#include <math.h>

// Problem constants (from reference)
#define NVERT 10242
#define DLEV  4
#define DIM   256
#define HEADS 8
#define HD    32
#define KNBR  8
#define MROWS (NVERT * DLEV)   // 40968
#define QKVN  (3 * DIM)        // 768

typedef __bf16 bf16_t;
typedef __bf16 v16bf __attribute__((ext_vector_type(16)));
typedef float  v8f   __attribute__((ext_vector_type(8)));

union BFrag { v16bf v; uint4 q[2]; };

// bf16 (packed in a dword) -> f32: bf16 is the high 16 bits of an f32.
__device__ inline float bflo(unsigned u) {
  unsigned v = u << 16; float f; __builtin_memcpy(&f, &v, 4); return f;
}
__device__ inline float bfhi(unsigned u) {
  unsigned v = u & 0xffff0000u; float f; __builtin_memcpy(&f, &v, 4); return f;
}

// ---------------------------------------------------------------------------
// Kernel 0: f32 -> bf16 weight conversion (one-time, tiny)
// ---------------------------------------------------------------------------
__global__ void cvt_f32_bf16(const float* __restrict__ src,
                             bf16_t* __restrict__ dst, int n) {
  int i = blockIdx.x * blockDim.x + threadIdx.x;
  if (i < n) dst[i] = (bf16_t)src[i];
}

// ---------------------------------------------------------------------------
// Kernel 1: QKV GEMM.  qkv[m, j] = sum_k x[m,k] * w[j,k] + b[j],
// q columns (j<256) additionally scaled by sqrt(HD) (reference: q / HD^-0.5).
// Wave computes a 16x64 strip via 4x v_wmma_f32_16x16x32_bf16 per K-step;
// K-loop fully unrolled (8 steps) so loads pipeline ahead of the WMMAs.
// A rows are CLAMPED (not predicated): out-of-range rows compute garbage in
// D rows that are never stored (WMMA rows are independent).
// ---------------------------------------------------------------------------
__global__ __launch_bounds__(128)
void qkv_gemm(const float* __restrict__ x,     // (MROWS, DIM) f32
              const bf16_t* __restrict__ w,    // (QKVN, DIM) bf16
              const float* __restrict__ bias,  // (QKVN,)
              bf16_t* __restrict__ qkv) {      // (MROWS, QKVN) bf16
  const int lane = threadIdx.x & 31;
  const int wave = threadIdx.x >> 5;
  const int r16  = lane & 15;
  const int hi   = lane >> 4;
  const int rowBase = (blockIdx.x * 4 + wave) * 16;
  const int colBase = blockIdx.y * 64;

  v8f c[4] = {};
  const int arow = min(rowBase + r16, MROWS - 1);  // branchless clamp
  const float* aPtr = x + (size_t)arow * DIM;

#pragma unroll
  for (int k0 = 0; k0 < DIM; k0 += 32) {
    // A fragment: lane<16 holds K[0..7]+K[16..23]; lane>=16 K[8..15]+K[24..31]
    BFrag a;
    const float4* p0 = (const float4*)(aPtr + k0 + hi * 8);
    const float4* p1 = (const float4*)(aPtr + k0 + 16 + hi * 8);
    float4 f0 = p0[0], f1 = p0[1], f2 = p1[0], f3 = p1[1];
    a.v = (v16bf){(bf16_t)f0.x, (bf16_t)f0.y, (bf16_t)f0.z, (bf16_t)f0.w,
                  (bf16_t)f1.x, (bf16_t)f1.y, (bf16_t)f1.z, (bf16_t)f1.w,
                  (bf16_t)f2.x, (bf16_t)f2.y, (bf16_t)f2.z, (bf16_t)f2.w,
                  (bf16_t)f3.x, (bf16_t)f3.y, (bf16_t)f3.z, (bf16_t)f3.w};
#pragma unroll
    for (int t = 0; t < 4; ++t) {
      const int col = colBase + t * 16 + r16;  // B column = weight row (W^T)
      BFrag b;
      const uint4* wp = (const uint4*)(w + (size_t)col * DIM + k0 + hi * 8);
      b.q[0] = wp[0]; b.q[1] = wp[2];          // K[hi*8..hi*8+7] + K[16+hi*8..16+hi*8+7]
      c[t] = __builtin_amdgcn_wmma_f32_16x16x32_bf16(
          false, a.v, false, b.v, (short)0, c[t], false, false);
    }
  }

#pragma unroll
  for (int t = 0; t < 4; ++t) {
    const int col = colBase + t * 16 + r16;
    const float bv  = bias[col];
    const float scl = (col < DIM) ? 5.656854249492381f : 1.0f;  // sqrt(32)
#pragma unroll
    for (int rr = 0; rr < 8; ++rr) {
      const int row = rowBase + rr + 8 * hi;   // D layout: M = rr + 8*(lane>=16)
      if (row < MROWS)
        qkv[(size_t)row * QKVN + col] = (bf16_t)((c[t][rr] + bv) * scl);
    }
  }
}

// ---------------------------------------------------------------------------
// Kernel 2: gathered neighborhood attention. One wave per (vertex n, head h).
// Lane j owns key/value j = (neighbor k=j/4, level e=j%4).  Logits are 32-wide
// bf16 dots in f32; softmax uses full-wave32 shfl_xor reductions; attn*V uses
// an LDS transpose (padded rows) to flip lanes to channel ownership.
// ---------------------------------------------------------------------------
__global__ __launch_bounds__(256)
void ico_attn(const bf16_t* __restrict__ qkv,       // (MROWS, 768)
              const int* __restrict__ which,        // (NVERT, 8)
              const unsigned char* __restrict__ mask,
              bf16_t* __restrict__ out) {           // (MROWS, 256)
  __shared__ unsigned v_sh[8][32 * 17];  // per-wave 32x32 bf16 V tile, padded rows
  __shared__ float    a_sh[8][4 * 32];   // per-wave 4x32 attention weights

  const int lane = threadIdx.x & 31;
  const int wave = threadIdx.x >> 5;     // = head h
  const int n = blockIdx.x;
  const int h = wave;
  const int kslot = lane >> 2;
  const int e     = lane & 3;
  const int  nb = which[n * KNBR + kslot];
  const bool mk = mask[n * KNBR + kslot] != 0;
  const size_t nbrow = (size_t)(nb * DLEV + e) * QKVN;

  // --- load this lane's key (32 bf16, contiguous, 64B aligned) ---
  const uint4* kp = (const uint4*)(qkv + nbrow + DIM + h * HD);
  uint4 kq[4]; kq[0] = kp[0]; kq[1] = kp[1]; kq[2] = kp[2]; kq[3] = kp[3];
  unsigned kw[16];
#pragma unroll
  for (int i = 0; i < 4; ++i) {
    kw[4*i+0] = kq[i].x; kw[4*i+1] = kq[i].y; kw[4*i+2] = kq[i].z; kw[4*i+3] = kq[i].w;
  }

  // --- logits: dot q[d,:] . key[:], q loads are wave-uniform (cache broadcast)
  float logit[4];
#pragma unroll
  for (int d = 0; d < 4; ++d) {
    const uint4* qp = (const uint4*)(qkv + (size_t)(n * DLEV + d) * QKVN + h * HD);
    uint4 qq[4]; qq[0] = qp[0]; qq[1] = qp[1]; qq[2] = qp[2]; qq[3] = qp[3];
    unsigned qw[16];
#pragma unroll
    for (int i = 0; i < 4; ++i) {
      qw[4*i+0] = qq[i].x; qw[4*i+1] = qq[i].y; qw[4*i+2] = qq[i].z; qw[4*i+3] = qq[i].w;
    }
    float acc = 0.f;
#pragma unroll
    for (int i = 0; i < 16; ++i)
      acc += bflo(qw[i]) * bflo(kw[i]) + bfhi(qw[i]) * bfhi(kw[i]);
    logit[d] = mk ? acc : -3.4028235e38f;  // reference masks with finfo.min
  }

  // --- softmax over 32 keys (one key per lane): full-wave reductions ---
  float attn[4];
#pragma unroll
  for (int d = 0; d < 4; ++d) {
    float m = logit[d];
#pragma unroll
    for (int off = 16; off > 0; off >>= 1) m = fmaxf(m, __shfl_xor(m, off, 32));
    float p = __expf(logit[d] - m);
    float s = p;
#pragma unroll
    for (int off = 16; off > 0; off >>= 1) s += __shfl_xor(s, off, 32);
    attn[d] = p / s;
  }

  // --- stage V tile (packed bf16) and attention weights into LDS ---
  const uint4* vp = (const uint4*)(qkv + nbrow + 2 * DIM + h * HD);
  uint4 vq[4]; vq[0] = vp[0]; vq[1] = vp[1]; vq[2] = vp[2]; vq[3] = vp[3];
  unsigned* vrow = &v_sh[wave][lane * 17];  // stride 17 dwords: conflict-free
#pragma unroll
  for (int i = 0; i < 4; ++i) {
    vrow[4*i+0] = vq[i].x; vrow[4*i+1] = vq[i].y; vrow[4*i+2] = vq[i].z; vrow[4*i+3] = vq[i].w;
  }
#pragma unroll
  for (int d = 0; d < 4; ++d) a_sh[wave][d * 32 + lane] = attn[d];
  __syncthreads();

  // --- flip to channel ownership: lane c computes y[d][c] = sum_j a[d][j]*v[j][c]
  const unsigned* vbase = &v_sh[wave][0];
  const float*    abase = &a_sh[wave][0];
  const int  cd = lane >> 1;       // dword holding this lane's channel
  const bool ch_hi = lane & 1;
  float acc[4] = {0.f, 0.f, 0.f, 0.f};
  for (int j = 0; j < 32; ++j) {
    const unsigned vv = vbase[j * 17 + cd];
    const float vf = ch_hi ? bfhi(vv) : bflo(vv);
#pragma unroll
    for (int d = 0; d < 4; ++d) acc[d] += abase[d * 32 + j] * vf;
  }
#pragma unroll
  for (int d = 0; d < 4; ++d)
    out[(size_t)(n * DLEV + d) * DIM + h * HD + lane] = (bf16_t)acc[d];
}

// ---------------------------------------------------------------------------
// Kernel 3: projection GEMM. out[m,j] = sum_k y[m,k]*pw[j,k] + pb[j], f32 out.
// Same WMMA structure; A is already bf16 in the workspace.
// ---------------------------------------------------------------------------
__global__ __launch_bounds__(128)
void proj_gemm(const bf16_t* __restrict__ a,    // (MROWS, DIM) bf16
               const bf16_t* __restrict__ w,    // (DIM, DIM) bf16
               const float* __restrict__ bias,  // (DIM,)
               float* __restrict__ out) {       // (MROWS, DIM) f32
  const int lane = threadIdx.x & 31;
  const int wave = threadIdx.x >> 5;
  const int r16  = lane & 15;
  const int hi   = lane >> 4;
  const int rowBase = (blockIdx.x * 4 + wave) * 16;
  const int colBase = blockIdx.y * 64;

  v8f c[4] = {};
  const int arow = min(rowBase + r16, MROWS - 1);  // branchless clamp
  const bf16_t* aPtr = a + (size_t)arow * DIM;

#pragma unroll
  for (int k0 = 0; k0 < DIM; k0 += 32) {
    BFrag af;
    af.q[0] = *(const uint4*)(aPtr + k0 + hi * 8);        // 8 bf16 = 16B
    af.q[1] = *(const uint4*)(aPtr + k0 + 16 + hi * 8);
#pragma unroll
    for (int t = 0; t < 4; ++t) {
      const int col = colBase + t * 16 + r16;
      BFrag b;
      const uint4* wp = (const uint4*)(w + (size_t)col * DIM + k0 + hi * 8);
      b.q[0] = wp[0]; b.q[1] = wp[2];
      c[t] = __builtin_amdgcn_wmma_f32_16x16x32_bf16(
          false, af.v, false, b.v, (short)0, c[t], false, false);
    }
  }

#pragma unroll
  for (int t = 0; t < 4; ++t) {
    const int col = colBase + t * 16 + r16;
    const float bv = bias[col];
#pragma unroll
    for (int rr = 0; rr < 8; ++rr) {
      const int row = rowBase + rr + 8 * hi;
      if (row < MROWS) out[(size_t)row * DIM + col] = c[t][rr] + bv;
    }
  }
}

// ---------------------------------------------------------------------------
// Workspace layout (bytes):
//   [0)       qkv_bf16   : MROWS*768*2  = 62,926,848
//   [+]       attn_bf16  : MROWS*256*2  = 20,975,616
//   [+]       qkv_w bf16 : 768*256*2    =    393,216
//   [+]       proj_w bf16: 256*256*2    =    131,072
//   total ~= 84.4 MB
// ---------------------------------------------------------------------------
extern "C" void kernel_launch(void* const* d_in, const int* in_sizes, int n_in,
                              void* d_out, int out_size, void* d_ws, size_t ws_size,
                              hipStream_t stream) {
  const float*         x      = (const float*)d_in[0];
  const int*           which  = (const int*)d_in[1];
  const unsigned char* mask   = (const unsigned char*)d_in[2];
  const float*         qkv_w  = (const float*)d_in[3];
  const float*         qkv_b  = (const float*)d_in[4];
  const float*         proj_w = (const float*)d_in[5];
  const float*         proj_b = (const float*)d_in[6];
  float*               out    = (float*)d_out;

  char* ws = (char*)d_ws;
  bf16_t* qkv_ws  = (bf16_t*)ws;
  size_t off = (size_t)MROWS * QKVN * 2;
  bf16_t* attn_ws = (bf16_t*)(ws + off);
  off += (size_t)MROWS * DIM * 2;
  bf16_t* wq = (bf16_t*)(ws + off);
  off += (size_t)QKVN * DIM * 2;
  bf16_t* wp = (bf16_t*)(ws + off);

  cvt_f32_bf16<<<(QKVN * DIM + 255) / 256, 256, 0, stream>>>(qkv_w, wq, QKVN * DIM);
  cvt_f32_bf16<<<(DIM * DIM + 255) / 256, 256, 0, stream>>>(proj_w, wp, DIM * DIM);

  dim3 g1((MROWS + 63) / 64, QKVN / 64);
  qkv_gemm<<<g1, 128, 0, stream>>>(x, wq, qkv_b, qkv_ws);

  ico_attn<<<NVERT, 256, 0, stream>>>(qkv_ws, which, mask, attn_ws);

  dim3 g2((MROWS + 63) / 64, DIM / 64);
  proj_gemm<<<g2, 128, 0, stream>>>(attn_ws, wp, proj_b, out);
}
